// AttentionHead_14577119003356
// MI455X (gfx1250) — matrix operand order repaired and checked by reading, not compiled, not run on hardware
//
#include <hip/hip_runtime.h>
#include <math.h>

// ---------------------------------------------------------------------------
// AttentionHead fused bf16-WMMA implementation for gfx1250 (MI455X).
//   out = softmax((X2 Wq + bq)(X1 Wk + bk)^T / sqrt(64)) (X1 Wv + bv)
// Phase 1 (projections): HBM-bound, one pass over 134 MB fp32 activations.
// Phase 2 (attention):   WMMA-bound flash attention; K tile via TDM
//                        (tensor_load_to_lds), V tile via async-to-LDS DMA,
//                        double buffered, one barrier per 64-wide tile.
// ---------------------------------------------------------------------------

typedef __bf16 bf16_t;
typedef __attribute__((ext_vector_type(16))) __bf16 v16bf;
typedef __attribute__((ext_vector_type(8)))  __bf16 v8bf;
typedef __attribute__((ext_vector_type(8)))  float  v8f;
typedef __attribute__((ext_vector_type(4)))  float  v4f;
typedef __attribute__((ext_vector_type(4)))  unsigned int v4u;
typedef __attribute__((ext_vector_type(8)))  int v8i;
typedef __attribute__((ext_vector_type(4)))  int v4i;

#define EMB   1024
#define DK    64
#define BATCH 4
#define SEQ   4096
#define SKT   64                       // Sk tile width
#define LOG2E 1.44269504088896340736f

// ---- fragment loaders (per CDNA5 ISA 7.12.2 layouts, wave32) --------------

// 16x32 bf16 A fragment from a row-major matrix (row stride ld elements).
__device__ inline v16bf load_a_frag(const bf16_t* __restrict__ base, int ld) {
  const int lane = threadIdx.x & 31;
  const int row  = lane & 15;
  const int k0   = (lane & 16) ? 8 : 0;
  const bf16_t* p = base + (size_t)row * ld + k0;
  v8bf lo = *(const v8bf*)(p);        // k0 .. k0+7
  v8bf hi = *(const v8bf*)(p + 16);   // k0+16 .. k0+23
  v16bf r;
#pragma unroll
  for (int i = 0; i < 8; ++i) { r[i] = lo[i]; r[i + 8] = hi[i]; }
  return r;
}

// Same A fragment sourced from fp32 memory, converted to bf16.
__device__ inline v16bf load_a_frag_f32(const float* __restrict__ base, int ld) {
  const int lane = threadIdx.x & 31;
  const int row  = lane & 15;
  const int k0   = (lane & 16) ? 8 : 0;
  const float* p = base + (size_t)row * ld + k0;
  v4f a0 = *(const v4f*)(p);
  v4f a1 = *(const v4f*)(p + 4);
  v4f b0 = *(const v4f*)(p + 16);
  v4f b1 = *(const v4f*)(p + 20);
  v16bf r;
#pragma unroll
  for (int i = 0; i < 4; ++i) {
    r[i]      = (bf16_t)a0[i];
    r[i + 4]  = (bf16_t)a1[i];
    r[i + 8]  = (bf16_t)b0[i];
    r[i + 12] = (bf16_t)b1[i];
  }
  return r;
}

// 32x16 bf16 B fragment from an N-major buffer: baseT[n*ld + k].
__device__ inline v16bf load_b_frag(const bf16_t* baseT, int ld) {
  const int lane = threadIdx.x & 31;
  const int n    = lane & 15;
  const int k0   = (lane & 16) ? 8 : 0;
  const bf16_t* p = baseT + (size_t)n * ld + k0;
  v8bf lo = *(const v8bf*)(p);
  v8bf hi = *(const v8bf*)(p + 16);
  v16bf r;
#pragma unroll
  for (int i = 0; i < 8; ++i) { r[i] = lo[i]; r[i + 8] = hi[i]; }
  return r;
}

__device__ inline v8f wmma_bf16(v16bf a, v16bf b, v8f c) {
  return __builtin_amdgcn_wmma_f32_16x16x32_bf16(
      false, a, false, b, (short)0, c, false, false);
}

// Async DMA: 16 bytes/lane, global -> LDS, tracked by ASYNCcnt.
__device__ inline void async_copy_b128(unsigned lds_byte_off, const bf16_t* g) {
  asm volatile("global_load_async_to_lds_b128 %0, %1, off"
               :: "v"(lds_byte_off), "v"((unsigned long long)(uintptr_t)g)
               : "memory");
}
__device__ inline unsigned lds_addr_of(const void* p) {
  // generic pointer to LDS: low 32 bits are the workgroup-relative LDS address
  return (unsigned)(uintptr_t)p;
}

// TDM: one 2-D bf16 tile (tile_d0 x tile_d1 elements, row stride stride_d0
// elements) global -> LDS.  D# bitfields per CDNA5 ISA 8.3/8.4.
// This toolchain exposes the 6-arg builtin:
//   (uint32x4 g0, int32x8 g1, int32x4 g2, int32x4 g3, int32x8 gx, i32 cpol)
__device__ inline void tdm_load_2d(unsigned lds_byte, const void* gptr,
                                   unsigned tile_d0, unsigned tile_d1,
                                   unsigned long long stride_d0) {
  unsigned long long ga = (unsigned long long)(uintptr_t)gptr;
  v4u g0;
  g0[0] = 1u;                                    // count=1, user mode
  g0[1] = lds_byte;                              // lds_addr
  g0[2] = (unsigned)(ga & 0xffffffffu);          // global_addr[31:0]
  g0[3] = (unsigned)((ga >> 32) & 0x01ffffffu)   // global_addr[56:32]
        | (2u << 30);                            // type = 2 ("image")
  v8i g1;
  g1[0] = (int)(1u << 16);                       // wg_mask=0, data_size=1 (2B)
  g1[1] = (int)((tile_d0 & 0xffffu) << 16);      // tensor_dim0[15:0] @ bit48
  g1[2] = (int)((tile_d0 >> 16) | ((tile_d1 & 0xffffu) << 16)); // dim0 hi|dim1 lo
  g1[3] = (int)((tile_d1 >> 16) | (tile_d0 << 16));             // dim1 hi|tile_dim0
  g1[4] = (int)(tile_d1 & 0xffffu);              // tile_dim1 | tile_dim2=0
  g1[5] = (int)(unsigned)(stride_d0 & 0xffffffffu);   // dim0_stride[31:0]
  g1[6] = (int)(unsigned)((stride_d0 >> 32) & 0xffffu); // dim0_stride[47:32]
  g1[7] = 0;                                     // dim1_stride (unused, 2D)
  v4i z4 = {0, 0, 0, 0};
  v8i z8 = {0, 0, 0, 0, 0, 0, 0, 0};
  __builtin_amdgcn_tensor_load_to_lds(g0, g1, z4, z4, z8, 0);
}

// ---------------------------------------------------------------------------
// Kernel 1: transpose + bf16-convert the three weight matrices.
//   Wt[p][d][e] = (bf16) W_p[e][d]   (N-major for B-fragment loads)
// ---------------------------------------------------------------------------
__global__ void prep_weights(const float* __restrict__ Wq,
                             const float* __restrict__ Wk,
                             const float* __restrict__ Wv,
                             bf16_t* __restrict__ Wt) {
  int idx = blockIdx.x * blockDim.x + threadIdx.x;   // 3*64*1024 total
  int p = idx / (DK * EMB);
  int r = idx % (DK * EMB);
  int d = r / EMB;
  int e = r % EMB;
  const float* W = (p == 0) ? Wq : (p == 1) ? Wk : Wv;
  Wt[idx] = (bf16_t)W[e * DK + d];
}

// ---------------------------------------------------------------------------
// Kernel 2: fused QKV projection, 2x-unrolled ping-pong pipeline over k
// (distinct registers per stage -> no buffer-rotation moves).
// ---------------------------------------------------------------------------
__global__ __launch_bounds__(256)
void qkv_proj(const float* __restrict__ input1,
              const float* __restrict__ input2,
              const float* __restrict__ bq,
              const float* __restrict__ bk,
              const float* __restrict__ bv,
              const bf16_t* __restrict__ Wt,   // [3][64][1024]
              bf16_t* __restrict__ Qb,         // [B*S][64]
              bf16_t* __restrict__ Kb,         // [B*S][64]
              bf16_t* __restrict__ Vt) {       // [B][64][S]
  const int proj = blockIdx.y;                  // 0=Q(from X2), 1=K, 2=V
  const int wave = threadIdx.x >> 5;
  const int lane = threadIdx.x & 31;
  const int s0   = blockIdx.x * 128 + wave * 16;   // global row (B*S flat)

  const float* X    = (proj == 0) ? input2 : input1;
  const float* bias = (proj == 0) ? bq : (proj == 1) ? bk : bv;
  const bf16_t* Wp  = Wt + (size_t)proj * DK * EMB;
  const float* Xrow = X + (size_t)s0 * EMB;

  v8f acc[4] = {{}, {}, {}, {}};

  v16bf a0 = load_a_frag_f32(Xrow, EMB);
  v16bf b0[4];
#pragma unroll
  for (int nt = 0; nt < 4; ++nt)
    b0[nt] = load_b_frag(Wp + (size_t)(nt * 16) * EMB, EMB);

  for (int k = 0; k < EMB; k += 64) {
    // stage 1 prefetch (always valid: k+32 <= EMB-32)
    v16bf a1 = load_a_frag_f32(Xrow + k + 32, EMB);
    v16bf b1[4];
#pragma unroll
    for (int nt = 0; nt < 4; ++nt)
      b1[nt] = load_b_frag(Wp + (size_t)(nt * 16) * EMB + k + 32, EMB);
#pragma unroll
    for (int nt = 0; nt < 4; ++nt) acc[nt] = wmma_bf16(a0, b0[nt], acc[nt]);

    if (k + 64 < EMB) {      // stage 0 prefetch for next iteration
      a0 = load_a_frag_f32(Xrow + k + 64, EMB);
#pragma unroll
      for (int nt = 0; nt < 4; ++nt)
        b0[nt] = load_b_frag(Wp + (size_t)(nt * 16) * EMB + k + 64, EMB);
    }
#pragma unroll
    for (int nt = 0; nt < 4; ++nt) acc[nt] = wmma_bf16(a1, b1[nt], acc[nt]);
  }

  const int half8 = (lane & 16) ? 8 : 0;   // C layout: lanes>=16 hold rows M+8
  const int nlo   = lane & 15;
#pragma unroll
  for (int nt = 0; nt < 4; ++nt) {
    const float bia = bias[nt * 16 + nlo];
    if (proj == 2) {
      int bb = s0 / SEQ;
      int sl = (s0 % SEQ) + half8;
      v8bf pk;
#pragma unroll
      for (int e = 0; e < 8; ++e) pk[e] = (bf16_t)(acc[nt][e] + bia);
      *(v8bf*)(Vt + ((size_t)bb * DK + nt * 16 + nlo) * SEQ + sl) = pk;
    } else {
      bf16_t* O = (proj == 0) ? Qb : Kb;
#pragma unroll
      for (int e = 0; e < 8; ++e)
        O[(size_t)(s0 + e + half8) * DK + nt * 16 + nlo] =
            (bf16_t)(acc[nt][e] + bia);
    }
  }
}

// ---------------------------------------------------------------------------
// Kernel 3: flash attention.  K tile: TDM descriptor load (wave 0 issues,
// TENSORcnt).  V tile: per-lane async-to-LDS (ASYNCcnt).  Double buffered.
// Per tile: 16 v_wmma (8 QK^T + 8 PV) + online softmax on VALU (co-executes).
// ---------------------------------------------------------------------------
__global__ __launch_bounds__(256)
void flash_attn(const bf16_t* __restrict__ Qb,
                const bf16_t* __restrict__ Kb,
                const bf16_t* __restrict__ Vt,
                float* __restrict__ out) {
  __shared__ __align__(32) bf16_t Ks[2][SKT][DK];   // 2 x 8 KB
  __shared__ __align__(32) bf16_t Vs[2][DK][SKT];   // 2 x 8 KB
  __shared__ __align__(32) bf16_t Plds[8][16][DK];  // 16 KB, one tile/wave

  const int tid  = threadIdx.x;
  const int wave = tid >> 5;
  const int lane = tid & 31;
  const int b    = blockIdx.y;
  const int q0   = blockIdx.x * 128 + wave * 16;
  const size_t gq0 = (size_t)b * SEQ + q0;

  const bf16_t* Kbb = Kb + (size_t)b * SEQ * DK;
  const bf16_t* Vtb = Vt + (size_t)b * DK * SEQ;
  bf16_t* myP = &Plds[wave][0][0];

  auto issue_tile = [&](int sk0, int buf) {
    if (wave == 0) {                            // K tile: one TDM descriptor
      tdm_load_2d(lds_addr_of(&Ks[buf][0][0]), Kbb + (size_t)sk0 * DK,
                  /*tile_d0=*/DK, /*tile_d1=*/SKT, /*stride_d0=*/DK);
    }
    {                                           // V tile: 64 rows of 128 B
      int d = tid >> 2;
      int c = (tid & 3) * 16;
      const bf16_t* g = Vtb + (size_t)d * SEQ + sk0 + c;
      unsigned l0 = lds_addr_of(&Vs[buf][d][c]);
      async_copy_b128(l0, g);
      async_copy_b128(l0 + 16, g + 8);
    }
  };
  auto wait_dma = [&]() {
    asm volatile("s_wait_asynccnt 0x0" ::: "memory");
    __builtin_amdgcn_s_wait_tensorcnt(0);       // trivial for non-issuing waves
  };

  // Q fragments are loop-invariant: load once.
  v16bf qa0 = load_a_frag(Qb + gq0 * DK + 0, DK);
  v16bf qa1 = load_a_frag(Qb + gq0 * DK + 32, DK);

  v8f o[4] = {{}, {}, {}, {}};
  v8f m, l;
#pragma unroll
  for (int e = 0; e < 8; ++e) { m[e] = -1e30f; l[e] = 0.0f; }

  const int half8 = (lane & 16) ? 8 : 0;
  const int nlo   = lane & 15;

  issue_tile(0, 0);
  wait_dma();
  __syncthreads();

  for (int it = 0; it < SEQ / SKT; ++it) {
    const int cur = it & 1;
    if (it + 1 < SEQ / SKT) issue_tile((it + 1) * SKT, cur ^ 1);

    // ---- S = (Q K^T) * 1/sqrt(dk), B-fragments from LDS -----------------
    v8f s[4];
#pragma unroll
    for (int nt = 0; nt < 4; ++nt) {
      v16bf bk0 = load_b_frag(&Ks[cur][nt * 16][0], DK);
      v16bf bk1 = load_b_frag(&Ks[cur][nt * 16][32], DK);
      v8f c = {};
      c = wmma_bf16(qa0, bk0, c);
      c = wmma_bf16(qa1, bk1, c);
#pragma unroll
      for (int e = 0; e < 8; ++e) s[nt][e] = c[e] * 0.125f;
    }

    // ---- online softmax (row stats via 16-lane xor tree) ----------------
    v8f mt;
#pragma unroll
    for (int e = 0; e < 8; ++e)
      mt[e] = fmaxf(fmaxf(s[0][e], s[1][e]), fmaxf(s[2][e], s[3][e]));
#pragma unroll
    for (int e = 0; e < 8; ++e) {
      float x = mt[e];
#pragma unroll
      for (int off = 8; off; off >>= 1) x = fmaxf(x, __shfl_xor(x, off, 32));
      mt[e] = x;
    }

    v8f mnew, alpha;
#pragma unroll
    for (int e = 0; e < 8; ++e) {
      mnew[e]  = fmaxf(m[e], mt[e]);
      alpha[e] = exp2f((m[e] - mnew[e]) * LOG2E);
    }

    v8f p[4], rs;
#pragma unroll
    for (int e = 0; e < 8; ++e) rs[e] = 0.0f;
#pragma unroll
    for (int nt = 0; nt < 4; ++nt)
#pragma unroll
      for (int e = 0; e < 8; ++e) {
        p[nt][e] = exp2f((s[nt][e] - mnew[e]) * LOG2E);
        rs[e] += p[nt][e];
      }
#pragma unroll
    for (int e = 0; e < 8; ++e) {
      float x = rs[e];
#pragma unroll
      for (int off = 8; off; off >>= 1) x += __shfl_xor(x, off, 32);
      rs[e] = x;
    }
#pragma unroll
    for (int e = 0; e < 8; ++e) {
      l[e] = l[e] * alpha[e] + rs[e];
      m[e] = mnew[e];
    }
#pragma unroll
    for (int nt = 0; nt < 4; ++nt)
#pragma unroll
      for (int e = 0; e < 8; ++e) o[nt][e] *= alpha[e];

    // ---- C-layout P -> row-major LDS tile -> A-layout fragments ---------
#pragma unroll
    for (int nt = 0; nt < 4; ++nt)
#pragma unroll
      for (int e = 0; e < 8; ++e)
        myP[(size_t)(e + half8) * DK + nt * 16 + nlo] = (bf16_t)p[nt][e];

    asm volatile("s_wait_dscnt 0x0" ::: "memory");  // same-wave LDS RAW

    v16bf pa0 = load_a_frag(myP + 0, DK);
    v16bf pa1 = load_a_frag(myP + 32, DK);

    // ---- O += P V, B-fragments from LDS ---------------------------------
#pragma unroll
    for (int nt = 0; nt < 4; ++nt) {
      v16bf bv0 = load_b_frag(&Vs[cur][nt * 16][0], SKT);
      v16bf bv1 = load_b_frag(&Vs[cur][nt * 16][32], SKT);
      o[nt] = wmma_bf16(pa0, bv0, o[nt]);
      o[nt] = wmma_bf16(pa1, bv1, o[nt]);
    }

    // prefetch DMA (TDM + async) must land before anyone reads it next iter
    wait_dma();
    __syncthreads();
  }

  // ---- normalize and store fp32 output ---------------------------------
  v8f linv;
#pragma unroll
  for (int e = 0; e < 8; ++e) linv[e] = 1.0f / l[e];
#pragma unroll
  for (int nt = 0; nt < 4; ++nt)
#pragma unroll
    for (int e = 0; e < 8; ++e)
      out[(gq0 + e + half8) * DK + nt * 16 + nlo] = o[nt][e] * linv[e];
}

// ---------------------------------------------------------------------------
extern "C" void kernel_launch(void* const* d_in, const int* in_sizes, int n_in,
                              void* d_out, int out_size, void* d_ws, size_t ws_size,
                              hipStream_t stream) {
  const float* input1 = (const float*)d_in[0];
  const float* input2 = (const float*)d_in[1];
  const float* Wq     = (const float*)d_in[2];
  const float* bq     = (const float*)d_in[3];
  const float* Wk     = (const float*)d_in[4];
  const float* bk     = (const float*)d_in[5];
  const float* Wv     = (const float*)d_in[6];
  const float* bv     = (const float*)d_in[7];
  float* out = (float*)d_out;

  // workspace (bf16): Wt[3][64][1024] | Qb[B*S][64] | Kb[B*S][64] | Vt[B][64][S]
  bf16_t* Wt = (bf16_t*)d_ws;
  bf16_t* Qb = Wt + (size_t)3 * DK * EMB;
  bf16_t* Kb = Qb + (size_t)BATCH * SEQ * DK;
  bf16_t* Vt = Kb + (size_t)BATCH * SEQ * DK;

  hipLaunchKernelGGL(prep_weights, dim3((3 * DK * EMB) / 256), dim3(256), 0,
                     stream, Wq, Wk, Wv, Wt);
  hipLaunchKernelGGL(qkv_proj, dim3((BATCH * SEQ) / 128, 3), dim3(256), 0,
                     stream, input1, input2, bq, bk, bv, Wt, Qb, Kb, Vt);
  hipLaunchKernelGGL(flash_attn, dim3(SEQ / 128, BATCH), dim3(256), 0,
                     stream, Qb, Kb, Vt, out);
}
